// _BatchedNN_80041010528773
// MI455X (gfx1250) — matrix operand order repaired and checked by reading, not compiled, not run on hardware
//
#include <hip/hip_runtime.h>

// ---------------- problem dimensions (fixed by the reference) ----------------
#define BATCH 64     // B conformers  (GEMM N dim)
#define ATOMS 128    // A
#define MODELS 8     // M ensemble
#define DAEV 384     // D
#define OUT0 256
#define OUT2 192
#define OUT4 160

// LDS activation strides (K + 8 bf16 pad -> stays 16B aligned, spreads banks)
#define KP_IN0 (DAEV + 8)   // 392
#define KP_0   (OUT0 + 8)   // 264
#define KP_2   (OUT2 + 8)   // 200
#define KP_4   (OUT4 + 8)   // 168

typedef __attribute__((ext_vector_type(16))) __bf16 v16bf;
typedef __attribute__((ext_vector_type(8)))  float  v8f;

union Frag16 {           // 16 bf16 = 8 VGPRs, element e = e-th 16-bit slot
    v16bf v;
    unsigned u[8];
    uint4 q[2];
};

// f32 pair -> packed bf16 dword in ONE v_perm_b32 (truncating round).
// perm(b, a, sel): bytes 0-3 index into a, 4-7 into b (little-endian result).
__device__ __forceinline__ unsigned pk_bf16(float a, float b) {
    return __builtin_amdgcn_perm(__float_as_uint(b), __float_as_uint(a),
                                 0x07060302u);
}

__device__ __forceinline__ float celu01(float x) {
    // celu(x, 0.1) = x>0 ? x : 0.1*(exp(10x)-1)
    return x > 0.f ? x : 0.1f * (__expf(x * 10.0f) - 1.0f);
}

// One dense layer: out[o, b] = celu(W[o,:] @ in[:, b] + bias[o]) for all 64 b.
// in/out live in LDS as bf16, [col=b][k] with stride KPI/KPO elements.
// W (f32, row-major [O][K]) and bias (f32) stream straight from HBM, once.
template<int K, int KPI, int O, int KPO>
__device__ void run_layer(const unsigned short* __restrict__ lin,
                          unsigned short* __restrict__ lout,
                          const float* __restrict__ W,
                          const float* __restrict__ bias)
{
    const int tid  = threadIdx.x;
    const int lane = tid & 31;
    const int wave = tid >> 5;
    const int ln15 = lane & 15;
    const int hi   = (lane >> 4) & 1;      // upper half-wave?
    constexpr int NOT = O / 16;            // 16-row output tiles

    for (int ot = wave; ot < NOT; ot += 8) {
        v8f acc[4];
        #pragma unroll
        for (int bt = 0; bt < 4; ++bt) acc[bt] = v8f{};

        const int row = ot * 16 + ln15;                 // A-matrix row (M)
        const float* wr = W + (size_t)row * K;

        for (int kb = 0; kb < K; kb += 32) {
            // ---- A fragment (16x32 bf16): lanes 0-15 hold K {kb..+7, kb+16..+23},
            //      lanes 16-31 hold K {kb+8..+15, kb+24..+31} for the same 16 rows.
            const int ka = kb + hi * 8;
            const float4 a0 = *(const float4*)(wr + ka);
            const float4 a1 = *(const float4*)(wr + ka + 4);
            const float4 a2 = *(const float4*)(wr + ka + 16);
            const float4 a3 = *(const float4*)(wr + ka + 20);
            if (kb + 32 < K)  // keep next k-step's weight line in flight
                __builtin_prefetch(wr + kb + 32 + hi * 8, 0, 3);

            Frag16 af;
            af.u[0] = pk_bf16(a0.x, a0.y); af.u[1] = pk_bf16(a0.z, a0.w);
            af.u[2] = pk_bf16(a1.x, a1.y); af.u[3] = pk_bf16(a1.z, a1.w);
            af.u[4] = pk_bf16(a2.x, a2.y); af.u[5] = pk_bf16(a2.z, a2.w);
            af.u[6] = pk_bf16(a3.x, a3.y); af.u[7] = pk_bf16(a3.z, a3.w);

            // ---- B fragments (32x16 bf16): lane n<16 = col n, K {kb..+7, kb+16..+23};
            //      lane n+16 = col n, K {kb+8..+15, kb+24..+31}. Load all four first so a
            //      single dscnt wait covers them, then fire 4 WMMAs back to back
            //      (A fragment reused from registers -> weights read 1x only).
            const int kB = kb + hi * 8;
            Frag16 bf[4];
            #pragma unroll
            for (int bt = 0; bt < 4; ++bt) {
                const unsigned short* p = lin + (bt * 16 + ln15) * KPI + kB;
                bf[bt].q[0] = *(const uint4*)p;
                bf[bt].q[1] = *(const uint4*)(p + 16);
            }
            #pragma unroll
            for (int bt = 0; bt < 4; ++bt) {
                acc[bt] = __builtin_amdgcn_wmma_f32_16x16x32_bf16(
                    false, af.v, false, bf[bt].v, (short)0, acc[bt], false, false);
            }
        }

        // ---- epilogue: C layout: acc[bt][r] = out[ot*16 + r + 8*hi][bt*16 + ln15]
        const int obase = ot * 16 + hi * 8;
        const float4 bv0 = *(const float4*)(bias + obase);
        const float4 bv1 = *(const float4*)(bias + obase + 4);
        const float bb[8] = { bv0.x, bv0.y, bv0.z, bv0.w,
                              bv1.x, bv1.y, bv1.z, bv1.w };
        #pragma unroll
        for (int bt = 0; bt < 4; ++bt) {
            float r[8];
            #pragma unroll
            for (int e = 0; e < 8; ++e) r[e] = celu01(acc[bt][e] + bb[e]);
            uint4 st;
            st.x = pk_bf16(r[0], r[1]);
            st.y = pk_bf16(r[2], r[3]);
            st.z = pk_bf16(r[4], r[5]);
            st.w = pk_bf16(r[6], r[7]);
            *(uint4*)(lout + (bt * 16 + ln15) * KPO + obase) = st;
        }
    }
}

__global__ __launch_bounds__(256)
void _BatchedNN_ani_kernel(const float* __restrict__ aev,
                           const float* __restrict__ w0, const float* __restrict__ b0,
                           const float* __restrict__ w2, const float* __restrict__ b2,
                           const float* __restrict__ w4, const float* __restrict__ b4,
                           const float* __restrict__ w6, const float* __restrict__ b6,
                           float* __restrict__ energies)
{
    extern __shared__ unsigned short smem[];
    unsigned short* buf0 = smem;                    // 64 * 392 bf16 (50176 B)
    unsigned short* buf1 = smem + BATCH * KP_IN0;   // 64 * 264 bf16 (33792 B)

    const int blk = blockIdx.x;          // one workgroup per (atom, model)
    const int a   = blk >> 3;
    const int am  = blk;                 // a*MODELS + m
    const int tid = threadIdx.x;

    // Stage aev[:, a, :] into LDS as bf16, [b][i] (B-matrix layout for layer 0)
    for (int idx = tid; idx < BATCH * (DAEV / 2); idx += 256) {
        const int b  = idx / (DAEV / 2);
        const int i2 = idx % (DAEV / 2);
        const float2 v = *(const float2*)(aev + ((size_t)b * ATOMS + a) * DAEV + 2 * i2);
        *(unsigned*)&buf0[b * KP_IN0 + 2 * i2] = pk_bf16(v.x, v.y);
    }
    __syncthreads();

    run_layer<DAEV, KP_IN0, OUT0, KP_0>(buf0, buf1,
        w0 + (size_t)am * OUT0 * DAEV, b0 + (size_t)am * OUT0);
    __syncthreads();
    run_layer<OUT0, KP_0, OUT2, KP_2>(buf1, buf0,
        w2 + (size_t)am * OUT2 * OUT0, b2 + (size_t)am * OUT2);
    __syncthreads();
    run_layer<OUT2, KP_2, OUT4, KP_4>(buf0, buf1,
        w4 + (size_t)am * OUT4 * OUT2, b4 + (size_t)am * OUT4);
    __syncthreads();

    // Final 1x160 layer + ensemble mean: e[b] += (w6 . x4[b] + b6) / M.
    // x4 (bf16) sits in buf1; buf0 region is free -> per-conformer partials.
    float* ered = (float*)smem;          // 64 floats in the dead buf0 region
    if (tid < BATCH) ered[tid] = 0.f;
    __syncthreads();

    {
        const int b = tid & 63;          // conformer
        const int q = tid >> 6;          // quarter: 40 of 160 elements
        const float* wf = w6 + (size_t)am * OUT4 + q * 40;
        const unsigned short* xp = buf1 + b * KP_4 + q * 40;  // 40 bf16, 4B aligned
        float s = 0.f;
        #pragma unroll
        for (int i = 0; i < 10; ++i) {   // uint2 = 4 bf16 per step
            const uint2 u = *(const uint2*)(xp + 4 * i);
            const float4 wv = *(const float4*)(wf + 4 * i);
            s += wv.x * __uint_as_float(u.x << 16);
            s += wv.y * __uint_as_float(u.x & 0xFFFF0000u);
            s += wv.z * __uint_as_float(u.y << 16);
            s += wv.w * __uint_as_float(u.y & 0xFFFF0000u);
        }
        atomicAdd(&ered[b], s);          // ds_add_f32
    }
    __syncthreads();

    if (tid < BATCH)
        atomicAdd(&energies[tid], (ered[tid] + b6[am]) * (1.0f / MODELS));
}

extern "C" void kernel_launch(void* const* d_in, const int* in_sizes, int n_in,
                              void* d_out, int out_size, void* d_ws, size_t ws_size,
                              hipStream_t stream) {
    const float* aev = (const float*)d_in[1];
    const float* w0  = (const float*)d_in[2];
    const float* b0  = (const float*)d_in[3];
    const float* w2  = (const float*)d_in[4];
    const float* b2  = (const float*)d_in[5];
    const float* w4  = (const float*)d_in[6];
    const float* b4  = (const float*)d_in[7];
    const float* w6  = (const float*)d_in[8];
    const float* b6  = (const float*)d_in[9];

    // Output tuple = (species passthrough [B*A i32], energies [B f32]) flat.
    hipMemcpyAsync(d_out, d_in[0], (size_t)BATCH * ATOMS * sizeof(int),
                   hipMemcpyDeviceToDevice, stream);
    float* energies = (float*)d_out + (size_t)BATCH * ATOMS;
    hipMemsetAsync(energies, 0, BATCH * sizeof(float), stream);

    const size_t lds_bytes = (size_t)(BATCH * KP_IN0 + BATCH * KP_0) * sizeof(unsigned short);
    _BatchedNN_ani_kernel<<<dim3(ATOMS * MODELS), dim3(256), lds_bytes, stream>>>(
        aev, w0, b0, w2, b2, w4, b4, w6, b6, energies);
}
